// Model_6562710028453
// MI455X (gfx1250) — matrix operand order repaired and checked by reading, not compiled, not run on hardware
//
#include <hip/hip_runtime.h>

typedef __attribute__((ext_vector_type(16))) _Float16 v16h;
typedef __attribute__((ext_vector_type(8)))  _Float16 v8h;
typedef __attribute__((ext_vector_type(8)))  float    v8f;
typedef __attribute__((ext_vector_type(4)))  unsigned int v4u;
typedef __attribute__((ext_vector_type(8)))  int v8i;
typedef __attribute__((ext_vector_type(4)))  int v4i;

__device__ __forceinline__ v16h cat16(v8h lo, v8h hi) {
  return __builtin_shufflevector(lo, hi, 0,1,2,3,4,5,6,7,8,9,10,11,12,13,14,15);
}

// CDNA5 16-bit A 16x32 fragment: lane half h holds K = h*8+0..7 (VGPR0-3) and
// K = 16+h*8+0..7 (VGPR4-7), row m = lane&15. Row-major LDS -> two b128 loads.
__device__ __forceinline__ v16h loadA(const _Float16* row, int hf) {
  const v8h* p = (const v8h*)(row + hf * 8);
  return cat16(p[0], p[2]);            // +0 and +16 elements
}
// B 32x16 fragment: lane half h holds K = h*8+0..7 and K = 16+h*8+0..7, col n = lane&15.
__device__ __forceinline__ v16h loadB(const _Float16* row, int hf) {
  const v8h* p = (const v8h*)(row + hf * 8);
  return cat16(p[0], p[2]);            // +0 and +16 elements
}

__global__ __launch_bounds__(256, 1) void nca_mlp_kernel(
    const float* __restrict__ state, const float* __restrict__ rnd,
    const float* __restrict__ w1g, const float* __restrict__ b1g,
    const float* __restrict__ w2g, const float* __restrict__ b2g,
    float* __restrict__ out, float* __restrict__ alpha_ws)
{
  __shared__ float                 s_state[16][18][18]; // 16ch 18x18 halo (TDM layout)
  __shared__ alignas(16) _Float16  s_perc[256][72];     // 256 px x 64 feats (+pad)
  __shared__ alignas(16) _Float16  s_w1[128][72];       // w1[o][k], k padded to 64
  __shared__ alignas(16) _Float16  s_w2[16][136];       // w2[p][k]
  __shared__ float                 s_b1[128];
  __shared__ float                 s_b2[16];
  __shared__ alignas(16) _Float16  s_x[8][16][136];     // per-wave hidden acts
  __shared__ float                 s_wout[8][16][17];   // per-wave [p][m] transpose

  const int tid = threadIdx.x;
  const int bb  = blockIdx.z;
  const int ty  = blockIdx.y, tx = blockIdx.x;
  const int wv  = tid >> 5;

  const bool interior = (tx > 0) && (tx < 15) && (ty > 0) && (ty < 15);

  // ---- halo staging: interior tiles via one Tensor-Data-Mover DMA ----
  if (interior) {
#if defined(__AMDGCN__)
    if (wv == 0) {
      // 3D tile: x=18, y=18, z=16 channels; elem=4B; strides 256 / 65536 elems
      size_t elem = (((size_t)bb * 16) * 256 + (ty * 16 - 1)) * 256 + (tx * 16 - 1);
      unsigned long long ga = (unsigned long long)(uintptr_t)state + elem * 4ull;
      unsigned lds = (unsigned)(uintptr_t)&s_state[0][0][0];
      v4u g0 = { 1u,                                   // count=1, user mode
                 lds,                                  // lds_addr
                 (unsigned)ga,                         // global_addr[31:0]
                 (unsigned)((ga >> 32) & 0x01FFFFFFu) | (2u << 30) }; // addr[56:32], type=2
      v8i g1 = { (int)(2u << 16),                      // data_size=4B, no mask/pad
                 (int)(18u << 16),                     // tensor_dim0[15:0]=18 in [63:48]
                 (int)(18u << 16),                     // dim0 hi=0 | tensor_dim1[15:0]=18
                 (int)(18u << 16),                     // dim1 hi=0 | tile_dim0=18
                 (int)(18u | (16u << 16)),             // tile_dim1=18, tile_dim2=16
                 (int)256,                             // tensor_dim0_stride lo32
                 0,                                    // stride0 hi | stride1[15:0]=0
                 (int)(65536u >> 16) };                // tensor_dim1_stride[47:16]=1
      v4i g2 = { 16, 0, 0, 0 };                        // tensor_dim2=16
      v4i g3 = { 0, 0, 0, 0 };
      v8i g4 = { 0, 0, 0, 0, 0, 0, 0, 0 };             // unused trailing group
      __builtin_amdgcn_tensor_load_to_lds(g0, g1, g2, g3, g4, 0);
      __builtin_amdgcn_s_wait_tensorcnt(0);
    }
#endif
  } else {
    for (int idx = tid; idx < 16 * 18 * 18; idx += 256) {
      int c = idx / 324, rem = idx - c * 324;
      int r = rem / 18, col = rem - r * 18;
      int gy = ty * 16 + r - 1, gx = tx * 16 + col - 1;
      float v = 0.f;
      if ((unsigned)gy < 256u && (unsigned)gx < 256u)
        v = state[(((size_t)(bb * 16 + c)) * 256 + gy) * 256 + gx];
      s_state[c][r][col] = v;
    }
  }

  // ---- stage weights (f32 -> f16) ----
  for (int idx = tid; idx < 128 * 64; idx += 256) {
    int o = idx >> 6, k = idx & 63;
    s_w1[o][k] = (k < 48) ? (_Float16)w1g[o * 48 + k] : (_Float16)0.f;
  }
  for (int idx = tid; idx < 16 * 128; idx += 256) {
    int p = idx >> 7, k = idx & 127;
    s_w2[p][k] = (_Float16)w2g[p * 128 + k];
  }
  if (tid < 128) s_b1[tid] = b1g[tid];
  if (tid < 16)  s_b2[tid] = b2g[tid];
  __syncthreads();

  // ---- depthwise sobel conv -> 48 f16 features per pixel (pad to 64) ----
  {
    int py = tid >> 4, px = tid & 15;
    _Float16* pr = &s_perc[tid][0];
#pragma unroll
    for (int c = 0; c < 16; ++c) {
      const float* sp = &s_state[c][py][px];
      float a00 = sp[0],  a01 = sp[1],  a02 = sp[2];
      float a10 = sp[18], a11 = sp[19], a12 = sp[20];
      float a20 = sp[36], a21 = sp[37], a22 = sp[38];
      float sx = (a02 - a00) + 2.f * (a12 - a10) + (a22 - a20);
      float sy = (a20 - a00) + 2.f * (a21 - a01) + (a22 - a02);
      pr[3 * c + 0] = (_Float16)sx;
      pr[3 * c + 1] = (_Float16)sy;
      pr[3 * c + 2] = (_Float16)a11;
    }
#pragma unroll
    for (int k = 48; k < 64; ++k) pr[k] = (_Float16)0.f;
  }
  __syncthreads();

  const int lane = tid & 31;
  const int hf   = lane >> 4;   // lane half
  const int mn   = lane & 15;   // M for A / N for B,C

#pragma unroll
  for (int gi = 0; gi < 2; ++gi) {            // each wave: 2 pixel rows
    const int g = wv * 2 + gi;
    const _Float16* prow = &s_perc[g * 16 + mn][0];
    v16h a0 = loadA(prow, hf);
    v16h a1 = loadA(prow + 32, hf);

    // ---- layer 1: 48(->64) -> 128, 8 N-tiles x 2 K-steps ----
#pragma unroll
    for (int t = 0; t < 8; ++t) {
      const _Float16* wrow = &s_w1[t * 16 + mn][0];
      v16h b0  = loadB(wrow, hf);
      v16h b1f = loadB(wrow + 32, hf);
      v8f acc = {0.f, 0.f, 0.f, 0.f, 0.f, 0.f, 0.f, 0.f};
      acc = __builtin_amdgcn_wmma_f32_16x16x32_f16(false, a0, false, b0,  (short)0, acc, false, false);
      acc = __builtin_amdgcn_wmma_f32_16x16x32_f16(false, a1, false, b1f, (short)0, acc, false, false);
      float bt = s_b1[t * 16 + mn];
#pragma unroll
      for (int r = 0; r < 8; ++r) {           // C: m = r + 8*half, n = mn
        float v = fmaxf(acc[r] + bt, 0.f);
        s_x[wv][r + 8 * hf][t * 16 + mn] = (_Float16)v;
      }
    }

    // ---- layer 2: 128 -> 16, 4 K-steps ----
    v8f acc2 = {0.f, 0.f, 0.f, 0.f, 0.f, 0.f, 0.f, 0.f};
#pragma unroll
    for (int q = 0; q < 4; ++q) {
      v16h ax = loadA(&s_x[wv][mn][0] + q * 32, hf);
      v16h bw = loadB(&s_w2[mn][0] + q * 32, hf);
      acc2 = __builtin_amdgcn_wmma_f32_16x16x32_f16(false, ax, false, bw, (short)0, acc2, false, false);
    }
#pragma unroll
    for (int r = 0; r < 8; ++r)
      s_wout[wv][mn][r + 8 * hf] = acc2[r];   // [p][m] transpose in LDS

    // ---- epilogue: state + ds*mask, coalesced (half-wave = 64B run) ----
    int gy = ty * 16 + g;
    int gxb = tx * 16;
#pragma unroll
    for (int it = 0; it < 8; ++it) {
      int p  = it * 2 + hf;
      int px = mn;
      float dsv = s_wout[wv][p][px] + s_b2[p];
      float st  = s_state[p][g + 1][px + 1];
      size_t idx = (((size_t)(bb * 16 + p)) * 256 + gy) * 256 + gxb + px;
      float rv = rnd[idx];
      float nv = st + ((rv < 0.5f) ? dsv : 0.f);
      out[idx] = nv;
      if (p == 3)
        alpha_ws[(((size_t)bb) * 256 + gy) * 256 + gxb + px] = nv;
    }
  }
}

// 3x3 max-pool on pre-mask alpha; zero all 16 channels of dead pixels.
__global__ __launch_bounds__(256, 4) void nca_alive_kernel(
    const float* __restrict__ alpha, float* __restrict__ out)
{
  int x = threadIdx.x;      // 0..255
  int y = blockIdx.x;       // 0..255
  int b = blockIdx.y;       // 0..15
  const float* ap = alpha + ((size_t)b) * 256 * 256;
  float m = -3.4e38f;
#pragma unroll
  for (int dy = -1; dy <= 1; ++dy) {
    int yy = y + dy;
    if ((unsigned)yy >= 256u) continue;
#pragma unroll
    for (int dx = -1; dx <= 1; ++dx) {
      int xx = x + dx;
      if ((unsigned)xx >= 256u) continue;
      m = fmaxf(m, ap[yy * 256 + xx]);
    }
  }
  if (!(m > 0.1f)) {
    size_t base = (((size_t)b * 16) * 256 + y) * 256 + x;
#pragma unroll
    for (int p = 0; p < 16; ++p)
      out[base + (size_t)p * 65536] = 0.f;
  }
}

extern "C" void kernel_launch(void* const* d_in, const int* in_sizes, int n_in,
                              void* d_out, int out_size, void* d_ws, size_t ws_size,
                              hipStream_t stream) {
  (void)in_sizes; (void)n_in; (void)out_size; (void)ws_size;
  const float* state  = (const float*)d_in[0];
  const float* rand_u = (const float*)d_in[1];
  // d_in[2] = conv_w: fixed sobel/identity filters, hardcoded in kernel
  const float* w1 = (const float*)d_in[3];
  const float* b1 = (const float*)d_in[4];
  const float* w2 = (const float*)d_in[5];
  const float* b2 = (const float*)d_in[6];
  float* out      = (float*)d_out;
  float* alpha_ws = (float*)d_ws;   // 16*256*256 f32 = 4 MB pre-mask alpha copy

  dim3 g1(16, 16, 16);              // (W/16, H/16, B)
  nca_mlp_kernel<<<g1, 256, 0, stream>>>(state, rand_u, w1, b1, w2, b2, out, alpha_ws);

  dim3 g2(256, 16);                 // (H, B), block = one W row
  nca_alive_kernel<<<g2, 256, 0, stream>>>(alpha_ws, out);
}
